// VxNet_22428319220767
// MI455X (gfx1250) — matrix operand order repaired and checked by reading, not compiled, not run on hardware
//
#include <hip/hip_runtime.h>

// =====================================================================
// Sparse 3D conv backbone (SECOND-style) for MI455X / gfx1250.
//
// Roofline: ~10 GFLOP total; all live data (grids 51MB + f16 features
// <16MB + weights ~2MB) fits in the 192MB L2 -> latency-bound, not
// HBM-bound. Strategy:
//  - f16 feature buffers; the 4-ch input is zero-padded to 16 ch so every
//    GEMM uses the vectorized CIN>=16 path (no scalar fallback at all)
//  - dense int32 occupancy grids; neighbor indices materialized once per
//    level and reused by the whole conv block
//  - one wave = 32 rows x cout: 2 M-tiles x NT N-tiles,
//    2*NT v_wmma_f32_16x16x32_f16 per 32-K chunk
//  - A-fragment 8-half runs are 8-aligned and never cross a tap for
//    CIN in {16,32,64,160}: 1 *unconditional* clamped nbidx load +
//    branchless pointer select (zrow) + global_load_b128 per run
//  - weights pre-swizzled once per layer into the exact WMMA-B lane
//    layout (f16) so B is two coalesced b128 loads per lane per chunk
//  - f16 epilogue stores are unguarded (buffers padded, pad rows never
//    read); only f32 stores to d_out/pw keep the row<n guard
// =====================================================================

#define BN_EPS 0.001f

typedef __attribute__((ext_vector_type(16))) _Float16 v16h;
typedef __attribute__((ext_vector_type(8)))  _Float16 v8h;
typedef __attribute__((ext_vector_type(8)))  float    v8f;

// ---------------- utility kernels ----------------
__global__ void fill_i32_kernel(int* __restrict__ p, long long n, int val) {
  long long i = (long long)blockIdx.x * blockDim.x + threadIdx.x;
  long long stride = (long long)gridDim.x * blockDim.x;
  for (; i < n; i += stride) p[i] = val;
}

__global__ void fill_f16_kernel(_Float16* __restrict__ p, int n) {
  int i = blockIdx.x * blockDim.x + threadIdx.x;
  if (i < n) p[i] = (_Float16)0.0f;
}

// 4-ch f32 voxel feats -> 16-ch zero-padded f16 rows (vector path everywhere)
__global__ void pad_feats_kernel(const float* __restrict__ in,
                                 _Float16* __restrict__ out, int n) {
  int t = blockIdx.x * blockDim.x + threadIdx.x;
  if (t >= n * 16) return;
  int i = t >> 4, c = t & 15;
  out[t] = (c < 4) ? (_Float16)in[i * 4 + c] : (_Float16)0.0f;
}

__global__ void build_grid_kernel(const int* __restrict__ coords, int n,
                                  int* __restrict__ grid, int D, int H, int W) {
  int i = blockIdx.x * blockDim.x + threadIdx.x;
  if (i >= n) return;
  const int* c = coords + 4 * i;
  grid[(((long long)c[0] * D + c[1]) * H + c[2]) * W + c[3]] = i;
}

// neighbor indices for one level: nbidx[i*27+tap] = grid[c*scale+off] or -1
__global__ void gather_nb_kernel(const int* __restrict__ coords, int n,
                                 const int* __restrict__ grid, int D, int H, int W,
                                 int scale, int* __restrict__ nbidx) {
  int t = blockIdx.x * blockDim.x + threadIdx.x;
  if (t >= n * 27) return;
  int i = t / 27, tap = t - i * 27;
  int dz = tap / 9 - 1, dy = (tap / 3) % 3 - 1, dx = tap % 3 - 1;
  const int* c = coords + 4 * i;
  int z = c[1] * scale + dz, y = c[2] * scale + dy, x = c[3] * scale + dx;
  int idx = -1;
  if (z >= 0 && z < D && y >= 0 && y < H && x >= 0 && x < W)
    idx = grid[(((long long)c[0] * D + z) * H + y) * W + x];
  nbidx[t] = idx;
}

// ---------------- weight pre-swizzle into WMMA-B lane layout ----------------
// Bs[((kc*NT+ct)*32 + lane)*16 + h] = W[tap*cin_src + c][ct*16 + (lane&15)]
// where k = kc*32 + (h&7) + 16*(h>>3) + 8*(lane>>4), tap = k/cin_dst, c = k%cin_dst, and the
// entry is zero when c >= cin_src (channel padding) or tap >= ntaps (K tail).
__global__ void stage_w_kernel(const float* __restrict__ W, _Float16* __restrict__ Bs,
                               int ntaps, int cin_src, int cin_dst, int cout,
                               int nt, int nelem) {
  int e = blockIdx.x * blockDim.x + threadIdx.x;
  if (e >= nelem) return;
  int h    = e & 15;
  int L    = (e >> 4) & 31;
  int rest = e >> 9;
  int ct   = rest % nt;
  int kc   = rest / nt;
  int k    = kc * 32 + (h & 7) + ((h >> 3) << 4) + ((L >> 4) << 3);
  int tap  = k / cin_dst;
  int c    = k - tap * cin_dst;
  int col  = ct * 16 + (L & 15);
  float v  = 0.0f;
  if (tap < ntaps && c < cin_src)
    v = W[((long long)tap * cin_src + c) * cout + col];
  Bs[e] = (_Float16)v;
}

// ---------------- register-blocked WMMA sparse-conv / GEMM ----------------
// OUTMODE: 0 = f16 out + BN/ReLU (unguarded store, padded buffer),
//          1 = f32 out + BN/ReLU (guarded), 2 = f32 raw (guarded).
template <int CIN, int NTAPS, int NT, int OUTMODE>
__global__ __launch_bounds__(32) void gemm_wmma_kernel(
    const _Float16* __restrict__ feats, const int* __restrict__ nbidx,
    const _Float16* __restrict__ Bs,
    const float* __restrict__ bn_g, const float* __restrict__ bn_b,
    const float* __restrict__ bn_m, const float* __restrict__ bn_v,
    void* __restrict__ outv, const _Float16* __restrict__ zrow, int n) {
  static_assert(CIN >= 16 && (CIN % 8) == 0, "vector A path requires CIN>=16");
  constexpr int Ktot = CIN * NTAPS;
  constexpr int NCH  = (Ktot + 31) / 32;
  constexpr int COUT = NT * 16;
  const int lane = threadIdx.x;
  const int mlo  = lane & 15;
  const int hi   = lane >> 4;
  const int row0 = blockIdx.x * 32;

  // clamped rows -> nbidx loads are always address-valid (branchless)
  int  rown[2], rowc[2];
  bool rok[2];
#pragma unroll
  for (int mt = 0; mt < 2; ++mt) {
    rown[mt] = row0 + mt * 16 + mlo;
    rok[mt]  = rown[mt] < n;
    rowc[mt] = rok[mt] ? rown[mt] : (n - 1);
  }

  v8f zeroacc = {};
  v8f acc[2][NT];
#pragma unroll
  for (int mt = 0; mt < 2; ++mt)
#pragma unroll
    for (int ct = 0; ct < NT; ++ct) acc[mt][ct] = zeroacc;

  for (int kc = 0; kc < NCH; ++kc) {
    // ---- A operands: 2 m-tiles x 2 runs, all unconditional b128 loads ----
    v16h a[2];
#pragma unroll
    for (int mt = 0; mt < 2; ++mt) {
#pragma unroll
      for (int part = 0; part < 2; ++part) {
        const int kstart = kc * 32 + part * 16 + hi * 8; // 8-aligned run
        const _Float16* p;
        if constexpr (NTAPS == 1) {
          bool ok = rok[mt] && (kstart < Ktot);
          p = ok ? (feats + (size_t)rown[mt] * CIN + kstart) : zrow;
        } else {
          const int tap = kstart / CIN;                  // compile-time shift
          const int c0  = kstart - tap * CIN;
          const int src = nbidx[rowc[mt] * 27 + tap];    // unconditional load
          bool ok = rok[mt] && (src >= 0) && (kstart < Ktot);
          p = ok ? (feats + (size_t)src * CIN + c0) : zrow; // v_cndmask pair
        }
        v8h seg = *(const v8h*)p;                        // global_load_b128
#pragma unroll
        for (int i = 0; i < 8; ++i) a[mt][part * 8 + i] = seg[i];
      }
    }
    // ---- B operands: pre-swizzled, coalesced b128 pairs ----
    v16h b[NT];
#pragma unroll
    for (int ct = 0; ct < NT; ++ct) {
      const _Float16* bp = Bs + ((size_t)(kc * NT + ct) * 32 + lane) * 16;
      v8h blo = *(const v8h*)bp;
      v8h bhi = *(const v8h*)(bp + 8);
#pragma unroll
      for (int i = 0; i < 8; ++i) { b[ct][i] = blo[i]; b[ct][8 + i] = bhi[i]; }
    }
    // ---- 2*NT WMMAs per chunk ----
#pragma unroll
    for (int ct = 0; ct < NT; ++ct)
#pragma unroll
      for (int mt = 0; mt < 2; ++mt)
        acc[mt][ct] = __builtin_amdgcn_wmma_f32_16x16x32_f16(
            false, a[mt], false, b[ct], (short)0, acc[mt][ct], false, false);
  }

  // ---- epilogue: fused BN + ReLU ----
#pragma unroll
  for (int ct = 0; ct < NT; ++ct) {
    const int col = ct * 16 + mlo;
    float scl = 1.0f, sft = 0.0f;
    if constexpr (OUTMODE != 2) {
      scl = bn_g[col] * rsqrtf(bn_v[col] + BN_EPS);
      sft = bn_b[col] - bn_m[col] * scl;
    }
#pragma unroll
    for (int mt = 0; mt < 2; ++mt) {
#pragma unroll
      for (int r = 0; r < 8; ++r) {
        const int row = row0 + mt * 16 + r + hi * 8;
        float v = acc[mt][ct][r];
        if constexpr (OUTMODE != 2) v = fmaxf(v * scl + sft, 0.0f);
        if constexpr (OUTMODE == 0) {
          // buffers padded to 64-row multiples; pad rows never consumed
          ((_Float16*)outv)[(size_t)row * COUT + col] = (_Float16)v;
        } else {
          if (row < n) ((float*)outv)[(size_t)row * COUT + col] = v;
        }
      }
    }
  }
}

// ---------------- KNN (k=3) inverse-distance interpolation ----------------
__global__ void knn_kernel(const float* __restrict__ pm, const int* __restrict__ coords,
                           const _Float16* __restrict__ feats, int nkeys, int cin,
                           float vs0, float vs1, float vs2,
                           _Float16* __restrict__ outp, int out_stride, int coloff) {
  const int pt = blockIdx.x;
  const int tid = threadIdx.x;
  const int NT = blockDim.x; // 128
  __shared__ float sd[3 * 128];
  __shared__ int   si[3 * 128];
  __shared__ float fw[3];
  __shared__ int   fi[3];
  float pb = pm[pt * 4 + 0];
  float px = pm[pt * 4 + 1], py = pm[pt * 4 + 2], pz = pm[pt * 4 + 3];
  float bd0 = 1e30f, bd1 = 1e30f, bd2 = 1e30f;
  int   bi0 = 0, bi1 = 0, bi2 = 0;
  for (int k = tid; k < nkeys; k += NT) {
    const int* c = coords + 4 * k;
    float kx = c[3] * vs0 + 0.5f * vs0;           // off.x = 0
    float ky = c[2] * vs1 - 40.0f + 0.5f * vs1;   // off.y = -40
    float kz = c[1] * vs2 - 3.0f + 0.5f * vs2;    // off.z = -3
    float dx = px - kx, dy = py - ky, dz = pz - kz;
    float d2 = dx * dx + dy * dy + dz * dz;
    if ((float)c[0] != pb) d2 = 1e10f;            // cross-batch mask
    if (d2 < bd2) {
      if (d2 < bd0)      { bd2=bd1; bi2=bi1; bd1=bd0; bi1=bi0; bd0=d2; bi0=k; }
      else if (d2 < bd1) { bd2=bd1; bi2=bi1; bd1=d2;  bi1=k; }
      else               { bd2=d2;  bi2=k; }
    }
  }
  sd[tid] = bd0; sd[NT + tid] = bd1; sd[2 * NT + tid] = bd2;
  si[tid] = bi0; si[NT + tid] = bi1; si[2 * NT + tid] = bi2;
  __syncthreads();
  if (tid == 0) {
    float m0 = 1e30f, m1 = 1e30f, m2 = 1e30f; int j0 = 0, j1 = 0, j2 = 0;
    for (int e = 0; e < 3 * NT; ++e) {
      float d = sd[e]; int id = si[e];
      if (d < m2) {
        if (d < m0)      { m2=m1;j2=j1; m1=m0;j1=j0; m0=d;j0=id; }
        else if (d < m1) { m2=m1;j2=j1; m1=d; j1=id; }
        else             { m2=d; j2=id; }
      }
    }
    float w0 = 1.0f/(m0+1e-8f), w1 = 1.0f/(m1+1e-8f), w2 = 1.0f/(m2+1e-8f);
    float wsum = w0 + w1 + w2;
    fw[0] = w0/wsum; fw[1] = w1/wsum; fw[2] = w2/wsum;
    fi[0] = j0; fi[1] = j1; fi[2] = j2;
  }
  __syncthreads();
  for (int cch = tid; cch < cin; cch += NT) {
    float v = fw[0] * (float)feats[(size_t)fi[0] * cin + cch]
            + fw[1] * (float)feats[(size_t)fi[1] * cin + cch]
            + fw[2] * (float)feats[(size_t)fi[2] * cin + cch];
    outp[(size_t)pt * out_stride + coloff + cch] = (_Float16)v;
  }
}

// ---------------- point heads (cls 64->1, reg 64->3) + pm copy ----------------
__global__ void heads_kernel(const float* __restrict__ pw,
                             const float* __restrict__ Wcls,
                             const float* __restrict__ Wreg,
                             const float* __restrict__ pm,
                             float* __restrict__ out_pm,
                             float* __restrict__ out_cls,
                             float* __restrict__ out_reg) {
  int pt = blockIdx.x;
  int tid = threadIdx.x; // 64
  __shared__ float s[64];
  s[tid] = pw[pt * 64 + tid];
  if (tid < 4) out_pm[pt * 4 + tid] = pm[pt * 4 + tid];
  __syncthreads();
  if (tid < 4) {
    float acc = 0.0f;
    if (tid == 0) {
      for (int c = 0; c < 64; ++c) acc += s[c] * Wcls[c];
      out_cls[pt] = acc;
    } else {
      int j = tid - 1;
      for (int c = 0; c < 64; ++c) acc += s[c] * Wreg[c * 3 + j];
      out_reg[pt * 3 + j] = acc;
    }
  }
}

// =====================================================================
// Host-side orchestration
// =====================================================================
extern "C" void kernel_launch(void* const* d_in, const int* in_sizes, int n_in,
                              void* d_out, int out_size, void* d_ws, size_t ws_size,
                              hipStream_t stream) {
  (void)n_in; (void)out_size; (void)ws_size;
  const float* voxel_feats = (const float*)d_in[0];
  const int*   coords0     = (const int*)d_in[1];
  const float* points_mean = (const float*)d_in[2];
  const int*   coords1     = (const int*)d_in[3];
  const int*   coords2     = (const int*)d_in[4];
  const int*   coords3     = (const int*)d_in[5];
  const int n0 = in_sizes[1] / 4;
  const int n1 = in_sizes[3] / 4;
  const int n2 = in_sizes[4] / 4;
  const int n3 = in_sizes[5] / 4;
  const int NP = in_sizes[2] / 4; // 2048 points

  // params pytree flattened in sorted-key order:
  // conv0[2], conv1[2], conv2[3], conv3[3], down0, down1, down2, extra,
  // point_cls, point_fc, point_reg; each conv dict as (W,b,g,m,v).
  struct Layer { const float *W, *b, *g, *m, *v; };
  int pi = 6;
  auto take_layer = [&](Layer& L) {
    L.W = (const float*)d_in[pi++]; L.b = (const float*)d_in[pi++];
    L.g = (const float*)d_in[pi++]; L.m = (const float*)d_in[pi++];
    L.v = (const float*)d_in[pi++];
  };
  Layer conv0[2], conv1[2], conv2[3], conv3[3], down0, down1, down2, extra;
  for (int i = 0; i < 2; ++i) take_layer(conv0[i]);
  for (int i = 0; i < 2; ++i) take_layer(conv1[i]);
  for (int i = 0; i < 3; ++i) take_layer(conv2[i]);
  for (int i = 0; i < 3; ++i) take_layer(conv3[i]);
  take_layer(down0); take_layer(down1); take_layer(down2); take_layer(extra);
  const float* Wcls = (const float*)d_in[pi++];
  const float* Wfc  = (const float*)d_in[pi++];
  const float* Wreg = (const float*)d_in[pi++];

  // ---- workspace bump allocator ----
  char* ws = (char*)d_ws; size_t wo = 0;
  auto walloc = [&](size_t bytes) -> void* {
    void* p = ws + wo; wo = (wo + bytes + 255) & ~(size_t)255; return p;
  };
  const long long G0 = 2LL * 40 * 400 * 352;
  const long long G1 = 2LL * 20 * 200 * 176;
  const long long G2 = 2LL * 10 * 100 * 88;
  const long long G3 = 2LL * 5 * 50 * 44;
  int* gall = (int*)walloc((size_t)(G0 + G1 + G2 + G3) * 4);
  int* g0 = gall;
  int* g1 = g0 + G0;
  int* g2 = g1 + G1;
  int* g3 = g2 + G2;
  int maxn = n0; if (n1 > maxn) maxn = n1; if (n2 > maxn) maxn = n2; if (n3 > maxn) maxn = n3;
  int maxnp = (maxn + 63) & ~63; // padded rows
  int*       nbbuf = (int*)walloc((size_t)maxnp * 27 * 4);
  _Float16*  vf16  = (_Float16*)walloc((size_t)n0 * 16 * 2);  // 16-ch padded input
  _Float16*  xa    = (_Float16*)walloc((size_t)maxnp * 64 * 2);
  _Float16*  xb    = (_Float16*)walloc((size_t)maxnp * 64 * 2);
  _Float16*  pcat  = (_Float16*)walloc((size_t)NP * 160 * 2);
  float*     pw    = (float*)walloc((size_t)NP * 64 * 4);
  _Float16*  zrow  = (_Float16*)walloc(256 * 2);              // zero A-row

  // weight swizzle buffers (f16, WMMA-B layout): nch*nt*512 halves each
  auto wsize = [](int Kdst, int cout) -> size_t {
    return (size_t)((Kdst + 31) / 32) * (cout / 16) * 512;
  };
  _Float16* Bs_c0a = (_Float16*)walloc(wsize(16 * 27, 16) * 2); // 4ch zero-padded
  _Float16* Bs_c0b = (_Float16*)walloc(wsize(16 * 27, 16) * 2);
  _Float16* Bs_d0  = (_Float16*)walloc(wsize(16 * 27, 32) * 2);
  _Float16* Bs_c1a = (_Float16*)walloc(wsize(32 * 27, 32) * 2);
  _Float16* Bs_c1b = (_Float16*)walloc(wsize(32 * 27, 32) * 2);
  _Float16* Bs_d1  = (_Float16*)walloc(wsize(32 * 27, 64) * 2);
  _Float16* Bs_c2a = (_Float16*)walloc(wsize(64 * 27, 64) * 2);
  _Float16* Bs_c2b = (_Float16*)walloc(wsize(64 * 27, 64) * 2);
  _Float16* Bs_c2c = (_Float16*)walloc(wsize(64 * 27, 64) * 2);
  _Float16* Bs_d2  = (_Float16*)walloc(wsize(64 * 27, 64) * 2);
  _Float16* Bs_c3a = (_Float16*)walloc(wsize(64 * 27, 64) * 2);
  _Float16* Bs_c3b = (_Float16*)walloc(wsize(64 * 27, 64) * 2);
  _Float16* Bs_c3c = (_Float16*)walloc(wsize(64 * 27, 64) * 2);
  _Float16* Bs_ex  = (_Float16*)walloc(wsize(64, 64) * 2);
  _Float16* Bs_fc  = (_Float16*)walloc(wsize(160, 64) * 2);

  // ---- output layout (tuple, concatenated flat, all f32) ----
  float* out_x   = (float*)d_out;                 // [n3, 64]
  float* out_pm  = out_x + (size_t)n3 * 64;       // [NP, 4]
  float* out_cls = out_pm + (size_t)NP * 4;       // [NP, 1]
  float* out_reg = out_cls + NP;                  // [NP, 3]

  auto stage_w = [&](const float* W, int ntaps, int cin_src, int cin_dst, int cout,
                     _Float16* Bs) {
    int nt = cout / 16, nch = (ntaps * cin_dst + 31) / 32, nelem = nch * nt * 512;
    stage_w_kernel<<<(nelem + 255) / 256, 256, 0, stream>>>(
        W, Bs, ntaps, cin_src, cin_dst, cout, nt, nelem);
  };
  auto gather = [&](const int* coords, int n, int* grid, int D, int H, int W, int scale) {
    gather_nb_kernel<<<(n * 27 + 255) / 256, 256, 0, stream>>>(
        coords, n, grid, D, H, W, scale, nbbuf);
  };

  // ---- independent prep: grids, weight swizzles, f16 conversions ----
  fill_i32_kernel<<<2048, 256, 0, stream>>>(gall, G0 + G1 + G2 + G3, -1);
  fill_f16_kernel<<<1, 256, 0, stream>>>(zrow, 256);
  pad_feats_kernel<<<(n0 * 16 + 255) / 256, 256, 0, stream>>>(voxel_feats, vf16, n0);
  build_grid_kernel<<<(n0 + 255) / 256, 256, 0, stream>>>(coords0, n0, g0, 40, 400, 352);
  build_grid_kernel<<<(n1 + 255) / 256, 256, 0, stream>>>(coords1, n1, g1, 20, 200, 176);
  build_grid_kernel<<<(n2 + 255) / 256, 256, 0, stream>>>(coords2, n2, g2, 10, 100, 88);
  build_grid_kernel<<<(n3 + 255) / 256, 256, 0, stream>>>(coords3, n3, g3, 5, 50, 44);
  stage_w(conv0[0].W, 27, 4, 16, 16, Bs_c0a);   // channel-padded 4->16
  stage_w(conv0[1].W, 27, 16, 16, 16, Bs_c0b);
  stage_w(down0.W, 27, 16, 16, 32, Bs_d0);
  stage_w(conv1[0].W, 27, 32, 32, 32, Bs_c1a);
  stage_w(conv1[1].W, 27, 32, 32, 32, Bs_c1b);
  stage_w(down1.W, 27, 32, 32, 64, Bs_d1);
  stage_w(conv2[0].W, 27, 64, 64, 64, Bs_c2a);
  stage_w(conv2[1].W, 27, 64, 64, 64, Bs_c2b);
  stage_w(conv2[2].W, 27, 64, 64, 64, Bs_c2c);
  stage_w(down2.W, 27, 64, 64, 64, Bs_d2);
  stage_w(conv3[0].W, 27, 64, 64, 64, Bs_c3a);
  stage_w(conv3[1].W, 27, 64, 64, 64, Bs_c3b);
  stage_w(conv3[2].W, 27, 64, 64, 64, Bs_c3c);
  stage_w(extra.W, 1, 64, 64, 64, Bs_ex);
  stage_w(Wfc, 1, 160, 160, 64, Bs_fc);

#define RUN_GEMM(CIN, NTAPS, NT, MODE, fin, nbi, Bs, L, fout, n)                      \
  gemm_wmma_kernel<CIN, NTAPS, NT, MODE><<<dim3(((n) + 31) / 32), 32, 0, stream>>>(   \
      fin, nbi, Bs, L.g, L.b, L.m, L.v, fout, zrow, n)

  // ---- level 0: conv0 (4(->16)->16, 16->16) ----
  gather(coords0, n0, g0, 40, 400, 352, 1);
  RUN_GEMM(16, 27, 1, 0, vf16, nbbuf, Bs_c0a, conv0[0], xa, n0);
  RUN_GEMM(16, 27, 1, 0, xa, nbbuf, Bs_c0b, conv0[1], xb, n0);
  // down0: 16->32 at coords1, gathering from grid0 at stride 2
  gather(coords1, n1, g0, 40, 400, 352, 2);
  RUN_GEMM(16, 27, 2, 0, xb, nbbuf, Bs_d0, down0, xa, n1);

  // ---- level 1: conv1 (32->32 x2), knn p1 ----
  gather(coords1, n1, g1, 20, 200, 176, 1);
  RUN_GEMM(32, 27, 2, 0, xa, nbbuf, Bs_c1a, conv1[0], xb, n1);
  RUN_GEMM(32, 27, 2, 0, xb, nbbuf, Bs_c1b, conv1[1], xa, n1);
  knn_kernel<<<NP, 128, 0, stream>>>(points_mean, coords1, xa, n1, 32,
                                     0.1f, 0.1f, 0.2f, pcat, 160, 0);
  // down1: 32->64 at coords2
  gather(coords2, n2, g1, 20, 200, 176, 2);
  RUN_GEMM(32, 27, 4, 0, xa, nbbuf, Bs_d1, down1, xb, n2);

  // ---- level 2: conv2 (64->64 x3), knn p2 ----
  gather(coords2, n2, g2, 10, 100, 88, 1);
  RUN_GEMM(64, 27, 4, 0, xb, nbbuf, Bs_c2a, conv2[0], xa, n2);
  RUN_GEMM(64, 27, 4, 0, xa, nbbuf, Bs_c2b, conv2[1], xb, n2);
  RUN_GEMM(64, 27, 4, 0, xb, nbbuf, Bs_c2c, conv2[2], xa, n2);
  knn_kernel<<<NP, 128, 0, stream>>>(points_mean, coords2, xa, n2, 64,
                                     0.2f, 0.2f, 0.4f, pcat, 160, 32);
  // down2: 64->64 at coords3
  gather(coords3, n3, g2, 10, 100, 88, 2);
  RUN_GEMM(64, 27, 4, 0, xa, nbbuf, Bs_d2, down2, xb, n3);

  // ---- level 3: conv3 (64->64 x3), knn p3 ----
  gather(coords3, n3, g3, 5, 50, 44, 1);
  RUN_GEMM(64, 27, 4, 0, xb, nbbuf, Bs_c3a, conv3[0], xa, n3);
  RUN_GEMM(64, 27, 4, 0, xa, nbbuf, Bs_c3b, conv3[1], xb, n3);
  RUN_GEMM(64, 27, 4, 0, xb, nbbuf, Bs_c3c, conv3[2], xa, n3);
  knn_kernel<<<NP, 128, 0, stream>>>(points_mean, coords3, xa, n3, 64,
                                     0.4f, 0.4f, 0.8f, pcat, 160, 96);

  // ---- heads ----
  // extra: 1x1 conv 64->64 + BN + ReLU (f32) straight into d_out
  RUN_GEMM(64, 1, 4, 1, xa, nullptr, Bs_ex, extra, out_x, n3);
  // point_fc: [NP,160] @ [160,64] dense WMMA GEMM, raw f32 out
  Layer fc; fc.W = Wfc; fc.b = fc.g = fc.m = fc.v = nullptr;
  RUN_GEMM(160, 1, 4, 2, pcat, nullptr, Bs_fc, fc, pw, NP);
  heads_kernel<<<NP, 64, 0, stream>>>(pw, Wcls, Wreg, points_mean,
                                      out_pm, out_cls, out_reg);
#undef RUN_GEMM
}
